// MoEBlock_57964878626956
// MI455X (gfx1250) — matrix operand order repaired and checked by reading, not compiled, not run on hardware
//
#include <hip/hip_runtime.h>
#include <math.h>

#define EMB     1024
#define HID     2816
#define NEXP    8
#define HCHUNK  128
#define NCHUNKS (HID / HCHUNK)   // 22
#define MTILE   32               // tokens per workgroup (2 WMMA M-tiles)
#define LDX     1026             // padded LDS stride (bf16 elems) for x tile
#define LDA     130              // padded LDS stride for activation tile

typedef __attribute__((ext_vector_type(16))) __bf16        v16bf;
typedef __attribute__((ext_vector_type(2)))  __bf16        v2bf;
typedef __attribute__((ext_vector_type(8)))  float         v8f;
typedef __attribute__((ext_vector_type(8)))  unsigned int  v8u;

// ---------- bf16 helpers: use HW converter (v_cvt_pk_bf16_f32) ----------
__device__ __forceinline__ unsigned int pack_bf16x2(float lo, float hi) {
    v2bf p;
    p[0] = (__bf16)lo;
    p[1] = (__bf16)hi;
    return __builtin_bit_cast(unsigned int, p);
}
__device__ __forceinline__ unsigned short f2bfu(float f) {
    __bf16 b = (__bf16)f;
    return __builtin_bit_cast(unsigned short, b);
}

// ---------- WMMA fragment loaders ----------
// A matrix (16xK bf16) from an LDS tile, row-major with stride ld (elements).
// Contiguous k order used here (A and B match): lane m = L&15; hi = L>>4; VGPR v holds K = 2*v + 16*hi (+0,+1)
// K-pairs are adjacent in LDS -> one ds_load_b32 per VGPR, no repacking.
__device__ __forceinline__ v16bf loadA_lds(const unsigned short* base, int ld,
                                           int kb, int lane) {
    const int m  = lane & 15;
    const int hi = lane >> 4;
    const unsigned short* row = base + m * ld + kb + (hi << 4);
    v8u a;
#pragma unroll
    for (int v = 0; v < 8; ++v) {
        const int k0 = 2 * v;
        a[v] = *(const unsigned int*)(row + k0);
    }
    return __builtin_bit_cast(v16bf, a);
}

// B matrix (32x16 slice of a KxN f32 row-major weight, stride ldw), cvt to bf16.
// Contiguous k order used here (A and B match): lane n = L&15; hi = L>>4; VGPR v holds K = 2*v + 16*hi (+0,+1)
__device__ __forceinline__ v16bf loadB_glb(const float* __restrict__ W, int ldw,
                                           int kb, int nb, int lane) {
    const int n  = lane & 15;
    const int hi = lane >> 4;
    const float* p = W + (size_t)(kb + (hi << 4)) * ldw + nb + n;
    v8u b;
#pragma unroll
    for (int v = 0; v < 8; ++v) {
        const float lo = p[(size_t)(2 * v) * ldw];
        const float hh = p[(size_t)(2 * v + 1) * ldw];
        b[v] = pack_bf16x2(lo, hh);
    }
    return __builtin_bit_cast(v16bf, b);
}

// ---------- kernel 0: zero output + counters ----------
__global__ void moe_zero_kernel(float* __restrict__ out, int n,
                                int* __restrict__ counts) {
    int i = blockIdx.x * blockDim.x + threadIdx.x;
    if (i < n) out[i] = 0.0f;
    if (i < NEXP) counts[i] = 0;
}

// ---------- kernel 1: router (scores -> top2 -> softmax -> scatter) ----------
__global__ __launch_bounds__(256) void moe_router_kernel(
        const float* __restrict__ x, const float* __restrict__ wr,
        int* __restrict__ counts, int* __restrict__ tokList,
        float* __restrict__ gateList, int T) {
    __shared__ float swr[EMB * NEXP];
    for (int i = threadIdx.x; i < EMB * NEXP; i += blockDim.x)
        swr[i] = wr[i];
    __syncthreads();

    const int t = blockIdx.x * blockDim.x + threadIdx.x;
    if (t >= T) return;

    float acc[NEXP];
#pragma unroll
    for (int e = 0; e < NEXP; ++e) acc[e] = 0.0f;

    const float* xr = x + (size_t)t * EMB;
    for (int d = 0; d < EMB; ++d) {
        const float xv = xr[d];
#pragma unroll
        for (int e = 0; e < NEXP; ++e)
            acc[e] = fmaf(xv, swr[d * NEXP + e], acc[e]);
    }

    // top-2 (distinct indices)
    int i0 = 0; float v0 = acc[0];
#pragma unroll
    for (int e = 1; e < NEXP; ++e)
        if (acc[e] > v0) { v0 = acc[e]; i0 = e; }
    int i1 = -1; float v1 = -3.4e38f;
#pragma unroll
    for (int e = 0; e < NEXP; ++e)
        if (e != i0 && acc[e] > v1) { v1 = acc[e]; i1 = e; }

    // softmax over {v0, v1} (v0 >= v1)
    const float e1  = __expf(v1 - v0);
    const float inv = __builtin_amdgcn_rcpf(1.0f + e1);
    const float g0  = inv;
    const float g1  = e1 * inv;

    int p0 = atomicAdd(&counts[i0], 1);
    tokList[(size_t)i0 * T + p0]  = t;
    gateList[(size_t)i0 * T + p0] = g0;
    int p1 = atomicAdd(&counts[i1], 1);
    tokList[(size_t)i1 * T + p1]  = t;
    gateList[(size_t)i1 * T + p1] = g1;
}

// ---------- kernel 2: fused grouped SwiGLU expert GEMM ----------
// One workgroup = 32 tokens (2 M-tiles) of one expert. 8 waves (wave32).
// Every B fragment feeds 2 WMMAs -> halves VMEM-per-FLOP vs M=16.
__global__ __launch_bounds__(256) void moe_expert_kernel(
        const float* __restrict__ x,  const float* __restrict__ w1,
        const float* __restrict__ w3, const float* __restrict__ w2,
        const int* __restrict__ counts, const int* __restrict__ tokList,
        const float* __restrict__ gateList, float* __restrict__ out,
        int T, int tilesPer) {
    __shared__ unsigned short xs[MTILE * LDX];    // 32 x EMB token tile (bf16)
    __shared__ unsigned short acts[MTILE * LDA];  // 32 x HCHUNK acts (bf16)
    __shared__ int   stok[MTILE];
    __shared__ float sgate[MTILE];

    const int e    = blockIdx.x / tilesPer;
    const int tile = blockIdx.x % tilesPer;
    const int cnt  = counts[e];
    if (tile * MTILE >= cnt) return;              // uniform exit

    const int tid = threadIdx.x;
    if (tid < MTILE) {
        const int idx = tile * MTILE + tid;
        const bool ok = idx < cnt;
        stok[tid]  = ok ? tokList[(size_t)e * T + idx]  : 0;
        sgate[tid] = ok ? gateList[(size_t)e * T + idx] : 0.0f;
    }
    __syncthreads();

    // gather 32 token rows into LDS as bf16 (b64 load + pack + b32 store)
    for (int i = tid; i < MTILE * (EMB / 2); i += 256) {
        const int r  = i >> 9;                  // EMB/2 == 512
        const int c2 = (i & 511) * 2;
        const float* row = x + (size_t)stok[r] * EMB + c2;
        *(unsigned int*)(xs + r * LDX + c2) = pack_bf16x2(row[0], row[1]);
    }
    __syncthreads();

    const int wave = tid >> 5;
    const int lane = tid & 31;
    const int n    = lane & 15;
    const int hi   = lane >> 4;

    const float* w1e = w1 + (size_t)e * EMB * HID;
    const float* w3e = w3 + (size_t)e * EMB * HID;
    const float* w2e = w2 + (size_t)e * HID * EMB;

    v8f yacc[2][8] = {};            // 2 M-tiles x 8 N-tiles of Y(32x1024)

    for (int hc = 0; hc < NCHUNKS; ++hc) {
        // ---- GEMM1: H/G (32 x 16 cols for this wave), K = EMB ----
        v8f hC[2] = {};
        v8f gC[2] = {};
        const int col = hc * HCHUNK + wave * 16;
        for (int kb = 0; kb < EMB; kb += 32) {
            const v16bf a0 = loadA_lds(xs, LDX, kb, lane);
            const v16bf a1 = loadA_lds(xs + 16 * LDX, LDX, kb, lane);
            const v16bf b1 = loadB_glb(w1e, HID, kb, col, lane);
            hC[0] = __builtin_amdgcn_wmma_f32_16x16x32_bf16(
                        false, a0, false, b1, (short)0, hC[0], false, false);
            hC[1] = __builtin_amdgcn_wmma_f32_16x16x32_bf16(
                        false, a1, false, b1, (short)0, hC[1], false, false);
            const v16bf b3 = loadB_glb(w3e, HID, kb, col, lane);
            gC[0] = __builtin_amdgcn_wmma_f32_16x16x32_bf16(
                        false, a0, false, b3, (short)0, gC[0], false, false);
            gC[1] = __builtin_amdgcn_wmma_f32_16x16x32_bf16(
                        false, a1, false, b3, (short)0, gC[1], false, false);
        }

        // ---- activation: silu(h) * g -> LDS (bf16), fast rcp ----
        __syncthreads();                // previous chunk's act reads done
#pragma unroll
        for (int mt = 0; mt < 2; ++mt) {
#pragma unroll
            for (int v = 0; v < 8; ++v) {
                const float h = hC[mt][v];
                const float s = h * __builtin_amdgcn_rcpf(1.0f + __expf(-h));
                acts[(mt * 16 + v + 8 * hi) * LDA + wave * 16 + n] =
                    f2bfu(s * gC[mt][v]);
            }
        }
        __syncthreads();

        // ---- GEMM2: Y += A(32x128) @ W2[chunk rows](128x1024) ----
        v16bf af[2][4];
#pragma unroll
        for (int mt = 0; mt < 2; ++mt)
#pragma unroll
            for (int kk = 0; kk < 4; ++kk)
                af[mt][kk] = loadA_lds(acts + mt * 16 * LDA, LDA, kk * 32, lane);

#pragma unroll
        for (int ntl = 0; ntl < 8; ++ntl) {
            const int nb = (wave * 8 + ntl) * 16;
#pragma unroll
            for (int kk = 0; kk < 4; ++kk) {
                const v16bf b = loadB_glb(w2e, EMB, hc * HCHUNK + kk * 32,
                                          nb, lane);
                yacc[0][ntl] = __builtin_amdgcn_wmma_f32_16x16x32_bf16(
                                   false, af[0][kk], false, b, (short)0,
                                   yacc[0][ntl], false, false);
                yacc[1][ntl] = __builtin_amdgcn_wmma_f32_16x16x32_bf16(
                                   false, af[1][kk], false, b, (short)0,
                                   yacc[1][ntl], false, false);
            }
        }
    }

    // ---- gate-scale and accumulate into output (each token has 2 experts) --
#pragma unroll
    for (int mt = 0; mt < 2; ++mt) {
#pragma unroll
        for (int ntl = 0; ntl < 8; ++ntl) {
            const int nb = (wave * 8 + ntl) * 16;
#pragma unroll
            for (int v = 0; v < 8; ++v) {
                const int m = mt * 16 + v + 8 * hi;
                const float val = yacc[mt][ntl][v] * sgate[m];
                unsafeAtomicAdd(&out[(size_t)stok[m] * EMB + nb + n], val);
            }
        }
    }
}

// ---------- host-side launcher ----------
extern "C" void kernel_launch(void* const* d_in, const int* in_sizes, int n_in,
                              void* d_out, int out_size, void* d_ws,
                              size_t ws_size, hipStream_t stream) {
    const float* x  = (const float*)d_in[0];
    const float* wr = (const float*)d_in[1];
    const float* w1 = (const float*)d_in[2];
    const float* w3 = (const float*)d_in[3];
    const float* w2 = (const float*)d_in[4];
    float* out = (float*)d_out;

    const int T = in_sizes[0] / EMB;            // tokens (4096)

    // workspace layout: counts | tokList[NEXP*T] | gateList[NEXP*T]
    char*  ws       = (char*)d_ws;
    int*   counts   = (int*)ws;
    int*   tokList  = (int*)(ws + 64);
    float* gateList = (float*)(ws + 64 + (size_t)NEXP * T * sizeof(int));

    moe_zero_kernel<<<(out_size + 255) / 256, 256, 0, stream>>>(out, out_size,
                                                                counts);
    moe_router_kernel<<<(T + 255) / 256, 256, 0, stream>>>(
        x, wr, counts, tokList, gateList, T);

    const int tilesPer = (T + MTILE - 1) / MTILE;
    moe_expert_kernel<<<NEXP * tilesPer, 256, 0, stream>>>(
        x, w1, w3, w2, counts, tokList, gateList, out, T, tilesPer);
}
